// GMMGCNLayer_39049842655442
// MI455X (gfx1250) — matrix operand order repaired and checked by reading, not compiled, not run on hardware
//
#include <hip/hip_runtime.h>
#include <math.h>

#define NN 4096
#define FIN 64
#define FOUT 64
#define KC 5
#define BB 8

typedef __attribute__((ext_vector_type(16))) __bf16 v16bf;
typedef __attribute__((ext_vector_type(8)))  float  v8f;

union V16U {
    v16bf v;
    uint4 q[2];
};

__device__ __forceinline__ unsigned short f2bf(float f) {
    unsigned u = __float_as_uint(f);
    unsigned r = u + 0x7FFFu + ((u >> 16) & 1u);   // round-to-nearest-even
    return (unsigned short)(r >> 16);
}

__device__ __forceinline__ float ex_relu_f(float mean, float var) {
    float v = fmaxf(var, 0.0f);
    float stdv = sqrtf(v);
    if (stdv > 0.0f) {
        float z = mean / stdv;
        float pdf = __expf(-0.5f * z * z) * 0.3989422804014327f;
        float cdf = 0.5f * (1.0f + erff(z * 0.7071067811865476f));
        return mean * cdf + stdv * pdf;
    }
    return fmaxf(mean, 0.0f);
}

// ---------------- Kernel 1: f32 -> bf16 conversion (shift, A2) ----------------
__global__ __launch_bounds__(256) void cvt_bf16_kernel(const float4* __restrict__ src,
                                                       ushort4* __restrict__ dst, int n4) {
    int idx = blockIdx.x * 256 + threadIdx.x;
    if (idx < n4) {
        float4 v = src[idx];
        ushort4 r;
        r.x = f2bf(v.x); r.y = f2bf(v.y); r.z = f2bf(v.z); r.w = f2bf(v.w);
        dst[idx] = r;
    }
}

// ---------------- Kernel 2: build txT/tcT (bf16, [b,k][o][m] layout) ----------
__global__ __launch_bounds__(256) void build_txtc_kernel(const float* __restrict__ feats,
                                                         const float* __restrict__ weight,
                                                         const float* __restrict__ mu,
                                                         const float* __restrict__ sigma,
                                                         unsigned short* __restrict__ txt,
                                                         unsigned short* __restrict__ tct) {
    const int mtile = blockIdx.x;   // 0..63  (64 rows of nodes each)
    const int k     = blockIdx.y;   // 0..4
    const int b     = blockIdx.z;   // 0..7
    const int t     = threadIdx.x;  // 0..255
    const int m0 = mtile * 64;

    __shared__ float sW[64 * 64];
    __shared__ float sF[64 * 64];
    __shared__ float sMu[64];
    __shared__ float sVar[64];
    __shared__ unsigned short sTx[64 * 65];  // [o][m_local]
    __shared__ unsigned short sTc[64 * 65];

    for (int i = t; i < 64 * 64; i += 256) {
        sW[i] = weight[i];
        sF[i] = feats[((size_t)b * NN + m0) * 64 + i];
    }
    if (t < 64) {
        sMu[t]  = mu[k * 64 + t];
        sVar[t] = __expf(sigma[k * 64 + t]);
    }
    __syncthreads();

    const int ml = t >> 2;       // local node row 0..63
    const int oq = t & 3;        // column quarter
    float txr[16], tcr[16];
#pragma unroll
    for (int j = 0; j < 16; ++j) { txr[j] = 0.0f; tcr[j] = 0.0f; }

    for (int f = 0; f < 64; ++f) {
        float x = sF[ml * 64 + f];
        bool nn = !(x == x);
        float mean = nn ? sMu[f] : x;
        float vv   = nn ? sVar[f] : 0.0f;
#pragma unroll
        for (int j = 0; j < 16; ++j) {
            float w = sW[f * 64 + oq * 16 + j];
            txr[j] += mean * w;
            tcr[j] += vv * (w * w);
        }
    }
#pragma unroll
    for (int j = 0; j < 16; ++j) {
        int o = oq * 16 + j;
        sTx[o * 65 + ml] = f2bf(txr[j]);
        sTc[o * 65 + ml] = f2bf(tcr[j]);
    }
    __syncthreads();

    // write out transposed: [ (b*K+k)*64 + o ][ m ] rows of length NN
    const int o  = t >> 2;       // 0..63
    const int ch = t & 3;        // chunk of 16 m's
    size_t rowbase = (((size_t)(b * KC + k)) * 64 + o) * NN + m0 + ch * 16;
#pragma unroll
    for (int i = 0; i < 16; ++i) {
        txt[rowbase + i] = sTx[o * 65 + ch * 16 + i];
        tct[rowbase + i] = sTc[o * 65 + ch * 16 + i];
    }
}

// ---------------- Kernel 3: responsibilities gamma[b,k,n] ---------------------
__global__ __launch_bounds__(256) void gamma_kernel(const float* __restrict__ feats,
                                                    const float* __restrict__ pi,
                                                    const float* __restrict__ mu,
                                                    const float* __restrict__ sigma,
                                                    float* __restrict__ gamma) {
    __shared__ float sMu[KC * 64];
    __shared__ float sRV[KC * 64];
    __shared__ float sPi[KC];
    const int t = threadIdx.x;
    for (int i = t; i < KC * 64; i += 256) {
        sMu[i] = mu[i];
        sRV[i] = __expf(-sigma[i]);   // 1/var
    }
    if (t < KC) sPi[t] = pi[t];
    __syncthreads();

    const int g = blockIdx.x * 256 + t;
    const int b = g >> 12;            // /4096
    const int n = g & (NN - 1);

    float acc[KC];
#pragma unroll
    for (int k = 0; k < KC; ++k) acc[k] = 0.0f;

    const float* frow = feats + ((size_t)b * NN + n) * 64;
    for (int f = 0; f < 64; ++f) {
        float x = frow[f];
        if (x == x) {
#pragma unroll
            for (int k = 0; k < KC; ++k) {
                float d = x - sMu[k * 64 + f];
                acc[k] += d * d * sRV[k * 64 + f];
            }
        }
    }
    float lg[KC], mx = -3.4e38f;
#pragma unroll
    for (int k = 0; k < KC; ++k) { lg[k] = sPi[k] - 0.5f * acc[k]; mx = fmaxf(mx, lg[k]); }
    float s = 0.0f;
#pragma unroll
    for (int k = 0; k < KC; ++k) { lg[k] = __expf(lg[k] - mx); s += lg[k]; }
    float inv = 1.0f / s;
#pragma unroll
    for (int k = 0; k < KC; ++k)
        gamma[((size_t)(b * KC + k)) * NN + n] = lg[k] * inv;
}

// ---------------- Kernel 4: fused WMMA GEMM + ex_relu + gamma-sum -------------
// Block: 256 threads = 8 waves, arranged 2 (row supertiles) x 4 (col tiles).
// Each wave computes a 32x16 output tile as two 16x16 WMMA tiles sharing every
// B fragment, for 5 components x {cx, cc} => 20 f32 accumulators per wave.
__global__ __launch_bounds__(256) void gmm_gcn_gemm_kernel(const unsigned short* __restrict__ shiftb,
                                                           const unsigned short* __restrict__ a2b,
                                                           const unsigned short* __restrict__ txt,
                                                           const unsigned short* __restrict__ tct,
                                                           const float* __restrict__ gamma,
                                                           float* __restrict__ out) {
    const int tid  = threadIdx.x;
    const int lane = tid & 31;
    const int wid  = tid >> 5;
    const int wr   = wid >> 2;            // 0..1 row supertile (32 rows each)
    const int wc   = wid & 3;             // 0..3 col tile (16 cols each)
    const int n0   = blockIdx.x * 64;     // block covers 64 rows
    const int b    = blockIdx.y;

    const int laneHi = lane >> 4;
    const int lm     = lane & 15;

    // A fragment addressing: row per lane; K chunks {0-7,16-23} / {8-15,24-31} by lane half
    const int arow0 = n0 + wr * 32 + lm;          // sub-tile 0 rows
    const unsigned short* aS0 = shiftb + (size_t)arow0 * NN + laneHi * 8;
    const unsigned short* aS1 = aS0 + (size_t)16 * NN;   // sub-tile 1 rows (+16)
    const unsigned short* aA0 = a2b + (size_t)arow0 * NN + laneHi * 8;
    const unsigned short* aA1 = aA0 + (size_t)16 * NN;

    // B fragment addressing: column per lane; K chunks {0-7,16-23} / {8-15,24-31} by lane half (txT rows)
    const int oG = wc * 16 + lm;
    const size_t brow = ((size_t)(b * KC)) * 64 + oG;    // first k's row index
    const unsigned short* pBbase = txt + brow * NN + laneHi * 8;
    const unsigned short* pCbase = tct + brow * NN + laneHi * 8;

    v8f accx0[KC], accx1[KC], accc0[KC], accc1[KC];
#pragma unroll
    for (int k = 0; k < KC; ++k) {
        accx0[k] = (v8f){0.f, 0.f, 0.f, 0.f, 0.f, 0.f, 0.f, 0.f};
        accx1[k] = (v8f){0.f, 0.f, 0.f, 0.f, 0.f, 0.f, 0.f, 0.f};
        accc0[k] = (v8f){0.f, 0.f, 0.f, 0.f, 0.f, 0.f, 0.f, 0.f};
        accc1[k] = (v8f){0.f, 0.f, 0.f, 0.f, 0.f, 0.f, 0.f, 0.f};
    }

#pragma unroll 1
    for (int mt = 0; mt < NN; mt += 32) {
        V16U a_s0, a_s1, a_a0, a_a1;
        a_s0.q[0] = *(const uint4*)(aS0 + mt);
        a_s0.q[1] = *(const uint4*)(aS0 + mt + 16);
        a_s1.q[0] = *(const uint4*)(aS1 + mt);
        a_s1.q[1] = *(const uint4*)(aS1 + mt + 16);
        a_a0.q[0] = *(const uint4*)(aA0 + mt);
        a_a0.q[1] = *(const uint4*)(aA0 + mt + 16);
        a_a1.q[0] = *(const uint4*)(aA1 + mt);
        a_a1.q[1] = *(const uint4*)(aA1 + mt + 16);

#pragma unroll
        for (int k = 0; k < KC; ++k) {
            const unsigned short* pB = pBbase + (size_t)k * 64 * NN + mt;
            V16U bx;
            bx.q[0] = *(const uint4*)(pB);
            bx.q[1] = *(const uint4*)(pB + 16);
            accx0[k] = __builtin_amdgcn_wmma_f32_16x16x32_bf16(
                false, a_s0.v, false, bx.v, (short)0, accx0[k], false, false);
            accx1[k] = __builtin_amdgcn_wmma_f32_16x16x32_bf16(
                false, a_s1.v, false, bx.v, (short)0, accx1[k], false, false);

            const unsigned short* pC = pCbase + (size_t)k * 64 * NN + mt;
            V16U bc;
            bc.q[0] = *(const uint4*)(pC);
            bc.q[1] = *(const uint4*)(pC + 16);
            accc0[k] = __builtin_amdgcn_wmma_f32_16x16x32_bf16(
                false, a_a0.v, false, bc.v, (short)0, accc0[k], false, false);
            accc1[k] = __builtin_amdgcn_wmma_f32_16x16x32_bf16(
                false, a_a1.v, false, bc.v, (short)0, accc1[k], false, false);
        }
    }

    // Epilogue: C/D layout — VGPR r: lanes 0-15 row M=r, lanes 16-31 row M=r+8
    const int rowBase0 = n0 + wr * 32 + laneHi * 8;      // sub-tile 0
    const int rowBase1 = rowBase0 + 16;                  // sub-tile 1
    float outv0[8], outv1[8];
#pragma unroll
    for (int r = 0; r < 8; ++r) { outv0[r] = 0.0f; outv1[r] = 0.0f; }

#pragma unroll
    for (int k = 0; k < KC; ++k) {
        const float* gp0 = gamma + ((size_t)(b * KC + k)) * NN + rowBase0;
        const float* gp1 = gp0 + 16;
#pragma unroll
        for (int r = 0; r < 8; ++r) {
            outv0[r] += gp0[r] * ex_relu_f(accx0[k][r], accc0[k][r]);
            outv1[r] += gp1[r] * ex_relu_f(accx1[k][r], accc1[k][r]);
        }
    }
#pragma unroll
    for (int r = 0; r < 8; ++r) {
        out[((size_t)b * NN + rowBase0 + r) * FOUT + oG] = outv0[r];
        out[((size_t)b * NN + rowBase1 + r) * FOUT + oG] = outv1[r];
    }
}

// ---------------- Launch ------------------------------------------------------
extern "C" void kernel_launch(void* const* d_in, const int* in_sizes, int n_in,
                              void* d_out, int out_size, void* d_ws, size_t ws_size,
                              hipStream_t stream) {
    const float* shift    = (const float*)d_in[0];   // [N,N]
    const float* A2       = (const float*)d_in[1];   // [N,N]
    const float* features = (const float*)d_in[2];   // [B,N,F]
    const float* weight   = (const float*)d_in[3];   // [F,O]
    const float* pi       = (const float*)d_in[4];   // [K]
    const float* mu       = (const float*)d_in[5];   // [K,F]
    const float* sigma    = (const float*)d_in[6];   // [K,F]
    float* out = (float*)d_out;                      // [B,N,O]

    // workspace carve-up (all offsets 256B aligned)
    unsigned short* shiftb = (unsigned short*)d_ws;                  // N*N bf16 = 32MB
    unsigned short* a2b    = shiftb + (size_t)NN * NN;               // 32MB
    unsigned short* txt    = a2b + (size_t)NN * NN;                  // B*K*64*N bf16 = 20MB
    unsigned short* tct    = txt + (size_t)BB * KC * 64 * NN;        // 20MB
    float*          gammap = (float*)(tct + (size_t)BB * KC * 64 * NN); // B*K*N f32

    // 1) f32 -> bf16 for the two adjacency operators
    {
        int n4 = NN * NN / 4;
        int blocks = (n4 + 255) / 256;
        cvt_bf16_kernel<<<blocks, 256, 0, stream>>>((const float4*)shift, (ushort4*)shiftb, n4);
        cvt_bf16_kernel<<<blocks, 256, 0, stream>>>((const float4*)A2,    (ushort4*)a2b,    n4);
    }
    // 2) txT / tcT
    {
        dim3 grid(NN / 64, KC, BB);
        build_txtc_kernel<<<grid, 256, 0, stream>>>(features, weight, mu, sigma, txt, tct);
    }
    // 3) gamma
    {
        int blocks = (BB * NN) / 256;
        gamma_kernel<<<blocks, 256, 0, stream>>>(features, pi, mu, sigma, gammap);
    }
    // 4) fused WMMA GEMM + epilogue
    {
        dim3 grid(NN / 64, BB);
        gmm_gcn_gemm_kernel<<<grid, 256, 0, stream>>>(shiftb, a2b, txt, tct, gammap, out);
    }
}
